// EncoderDecoderMaster_8143257993755
// MI455X (gfx1250) — matrix operand order repaired and checked by reading, not compiled, not run on hardware
//
#include <hip/hip_runtime.h>
#include <math.h>

// ---------------------------------------------------------------------------
// Types
// ---------------------------------------------------------------------------
typedef __bf16 bf16;
typedef __attribute__((ext_vector_type(16))) __bf16 bf16x16;
typedef __attribute__((ext_vector_type(8)))  float  floatx8;
typedef __attribute__((ext_vector_type(4)))  int    i32x4;

#define BATCH 64
#define SEQT  256
#define INP   256
#define HID   1024
#define G4    4096   // 4*HID

#if __has_builtin(__builtin_amdgcn_global_load_async_to_lds_b128) && \
    __has_builtin(__builtin_amdgcn_s_wait_asynccnt)
#define USE_ASYNC_LDS 1
#endif

__device__ __forceinline__ floatx8 wmma_bf16(bf16x16 a, bf16x16 b, floatx8 c) {
  return __builtin_amdgcn_wmma_f32_16x16x32_bf16(false, a, false, b, (short)0, c,
                                                 false, false);
}

// A-matrix fragment (16x32 bf16, MxK). Lane l holds row m = l%16.
// VGPR v (pair of bf16): k = (v/4)*16 + (l/16)*8 + (v%4)*2
__device__ __forceinline__ bf16x16 load_fragA(const bf16* base, int stride, int lane) {
  const bf16* p = base + (lane & 15) * stride;
  const int hh = (lane >> 4) * 8;
  bf16x16 f;
#pragma unroll
  for (int v = 0; v < 8; ++v) {
    const int k = ((v >> 2) << 4) + hh + ((v & 3) << 1);
    f[2 * v]     = p[k];
    f[2 * v + 1] = p[k + 1];
  }
  return f;
}

// B-matrix fragment (32x16 bf16, KxN), data stored as 16 rows (one per N col),
// contiguous in K with given stride. Lane l holds col n = l%16.
// VGPR v: k = (v/4)*16 + (l/16)*8 + (v%4)*2
__device__ __forceinline__ bf16x16 load_fragB(const bf16* base, int stride, int lane) {
  const bf16* p = base + (lane & 15) * stride;
  const int hh = (lane >> 4) * 8;
  bf16x16 f;
#pragma unroll
  for (int v = 0; v < 8; ++v) {
    const int k = ((v >> 2) << 4) + hh + ((v & 3) << 1);
    f[2 * v]     = p[k];
    f[2 * v + 1] = p[k + 1];
  }
  return f;
}

__device__ __forceinline__ float sigmoidf_(float x) {
  return 1.0f / (1.0f + __expf(-x));
}

// 16-byte global -> LDS transfer (async on CDNA5, VGPR-mediated fallback).
__device__ __forceinline__ void g2l_b128(const bf16* g, bf16* l) {
#ifdef USE_ASYNC_LDS
  __builtin_amdgcn_global_load_async_to_lds_b128(
      (__attribute__((address_space(1))) i32x4*)(unsigned long long)g,
      (__attribute__((address_space(3))) i32x4*)l, 0, 0);
#else
  *(uint4*)l = *(const uint4*)g;
#endif
}

__device__ __forceinline__ void wait_async_le20() {
#ifdef USE_ASYNC_LDS
  __builtin_amdgcn_s_wait_asynccnt(20);
#endif
}
__device__ __forceinline__ void wait_async_le0() {
#ifdef USE_ASYNC_LDS
  __builtin_amdgcn_s_wait_asynccnt(0);
#endif
}

// ---------------------------------------------------------------------------
// Utility kernels
// ---------------------------------------------------------------------------
__global__ void cvt_f32_to_bf16(const float* __restrict__ in,
                                bf16* __restrict__ out, int n) {
  int i = blockIdx.x * blockDim.x + threadIdx.x;
  if (i < n) out[i] = (bf16)in[i];
}

__global__ void zero_u32(unsigned int* __restrict__ p, int n) {
  int i = blockIdx.x * blockDim.x + threadIdx.x;
  if (i < n) p[i] = 0u;
}

// ---------------------------------------------------------------------------
// X-projection GEMM: P[t][b][n] = sum_k x[b][t][k] * Wih[n][k] + bih[n] + bhh[n]
// Grid: (T, 4096/128). Block: 256 threads (8 waves).
// Block tile: M=64 (batch), N=128, K=256 (full). Wave w owns N-tile w; 4 M-tiles.
// ---------------------------------------------------------------------------
__global__ void xproj_kernel(const bf16* __restrict__ xb,   // [64][256][256]
                             const bf16* __restrict__ wih,  // [4096][256]
                             const float* __restrict__ bih, // [4096]
                             const float* __restrict__ bhh, // [4096]
                             float* __restrict__ P)         // [256][64][4096]
{
  extern __shared__ char smem[];
  bf16* x_s = (bf16*)smem;                      // [64][256]  32 KB
  bf16* w_s = (bf16*)(smem + BATCH * INP * 2);  // [128][256] 64 KB

  const int tid  = threadIdx.x;
  const int lane = tid & 31;
  const int wave = tid >> 5;
  const int t    = blockIdx.x;
  const int n0   = blockIdx.y * 128;

  // Stage x rows for this t: 64 rows x 128 dword-pairs
  for (int u = tid; u < 64 * 128; u += 256) {
    int b  = u >> 7;
    int kk = u & 127;
    ((unsigned int*)x_s)[b * 128 + kk] =
        ((const unsigned int*)(xb + ((size_t)b * SEQT + t) * INP))[kk];
  }
  // Stage Wih slice: 128 rows x 128 dword-pairs
  for (int u = tid; u < 128 * 128; u += 256) {
    int r  = u >> 7;
    int kk = u & 127;
    ((unsigned int*)w_s)[r * 128 + kk] =
        ((const unsigned int*)(wih + (size_t)(n0 + r) * INP))[kk];
  }
  __syncthreads();

  floatx8 acc[4];
#pragma unroll
  for (int m = 0; m < 4; ++m)
#pragma unroll
    for (int e = 0; e < 8; ++e) acc[m][e] = 0.0f;

  const int nt = wave;  // N-tile (0..7)
#pragma unroll
  for (int ks = 0; ks < 8; ++ks) {
    bf16x16 bfrag = load_fragB(w_s + (nt * 16) * INP + ks * 32, INP, lane);
#pragma unroll
    for (int m = 0; m < 4; ++m) {
      bf16x16 afrag = load_fragA(x_s + (m * 16) * INP + ks * 32, INP, lane);
      acc[m] = wmma_bf16(afrag, bfrag, acc[m]);
    }
  }

  // Epilogue: add biases, store fp32
  const int nloc = lane & 15;
  const int mh   = (lane >> 4) * 8;
  const int n    = n0 + nt * 16 + nloc;
  const float bias = bih[n] + bhh[n];
#pragma unroll
  for (int m = 0; m < 4; ++m) {
#pragma unroll
    for (int j = 0; j < 8; ++j) {
      int b = m * 16 + mh + j;
      P[((size_t)t * BATCH + b) * G4 + n] = acc[m][j] + bias;
    }
  }
}

// ---------------------------------------------------------------------------
// One LSTM timestep. Grid: 16 blocks (each owns 64 h-columns and the matching
// 4x64 gate columns). Block: 256 threads (8 waves).
// gates[b][n] = P_t[b][n] + sum_k h_in[b][k] * Whh[n][k]   (K = 1024)
// LDS double-buffered, staged with async-to-LDS; fragments register-pipelined.
// ---------------------------------------------------------------------------
#define HCHUNK_B  (BATCH * 128 * 2)   // 16 KB  (h chunk [64][128] bf16)
#define WCHUNK_B  (256 * 128 * 2)     // 64 KB  (Whh chunk [256][128] bf16)
#define STAGE_OPS 20                  // async b128 ops per thread per chunk

// Issue the global->LDS staging for K-chunk kc into (h_s, w_s).
__device__ __forceinline__ void stage_chunk(const bf16* __restrict__ h_in,
                                            const bf16* __restrict__ whh,
                                            bf16* h_s, bf16* w_s,
                                            int k0, int n0, int tid) {
  // h chunk: 64 rows x 128 cols -> 1024 x 16B ops (16 per row)
#pragma unroll
  for (int j = 0; j < 4; ++j) {
    const int u   = tid + j * 256;
    const int row = u >> 4;
    const int seg = u & 15;
    g2l_b128(h_in + (size_t)row * HID + k0 + seg * 8, h_s + row * 128 + seg * 8);
  }
  // Whh chunk: 256 rows (r = gate*64 + col -> Whh row gate*1024 + n0 + col)
#pragma unroll
  for (int j = 0; j < 16; ++j) {
    const int u   = tid + j * 256;
    const int r   = u >> 4;
    const int seg = u & 15;
    const int g   = r >> 6;
    const int cc  = r & 63;
    g2l_b128(whh + (size_t)(g * HID + n0 + cc) * HID + k0 + seg * 8,
             w_s + r * 128 + seg * 8);
  }
}

__global__ void lstm_step_kernel(const bf16* __restrict__ h_in,  // [64][1024]
                                 bf16* __restrict__ h_out,       // [64][1024]
                                 float* __restrict__ cst,        // [64][1024]
                                 const bf16* __restrict__ whh,   // [4096][1024]
                                 const float* __restrict__ P_t)  // [64][4096]
{
  extern __shared__ char smem[];
  bf16* h_sb[2];
  bf16* w_sb[2];
  h_sb[0] = (bf16*)smem;
  w_sb[0] = (bf16*)(smem + HCHUNK_B);
  h_sb[1] = (bf16*)(smem + HCHUNK_B + WCHUNK_B);
  w_sb[1] = (bf16*)(smem + 2 * HCHUNK_B + WCHUNK_B);
  float* g_s = (float*)(smem + 2 * (HCHUNK_B + WCHUNK_B));  // [64][256] 64 KB

  const int tid  = threadIdx.x;
  const int lane = tid & 31;
  const int wave = tid >> 5;
  const int n0   = blockIdx.x * 64;  // h-column base

  floatx8 acc[4][2];
#pragma unroll
  for (int m = 0; m < 4; ++m)
#pragma unroll
    for (int i = 0; i < 2; ++i)
#pragma unroll
      for (int e = 0; e < 8; ++e) acc[m][i][e] = 0.0f;

  // Prime the pipeline: stage chunk 0 into buffer 0.
  stage_chunk(h_in, whh, h_sb[0], w_sb[0], 0, n0, tid);

  for (int kc = 0; kc < 8; ++kc) {
    const int cur = kc & 1;
    // Issue staging for the next chunk (prev compute barrier made buf safe).
    if (kc < 7) {
      stage_chunk(h_in, whh, h_sb[cur ^ 1], w_sb[cur ^ 1], (kc + 1) * 128, n0, tid);
      // Async loads complete in order: <= STAGE_OPS outstanding means the
      // current chunk's transfers have landed in LDS.
      wait_async_le20();
    } else {
      wait_async_le0();
    }
    __syncthreads();  // all waves' staging of `cur` visible

    const bf16* hb = h_sb[cur];
    const bf16* wb = w_sb[cur];

    // Register-pipelined fragment loads over the 4 k-steps of this chunk.
    bf16x16 af[2][4];
    bf16x16 bf_[2][2];
#pragma unroll
    for (int m = 0; m < 4; ++m)
      af[0][m] = load_fragA(hb + (m * 16) * 128, 128, lane);
#pragma unroll
    for (int i = 0; i < 2; ++i)
      bf_[0][i] = load_fragB(wb + ((wave * 2 + i) * 16) * 128, 128, lane);

#pragma unroll
    for (int ks = 0; ks < 4; ++ks) {
      const int cb = ks & 1;
      const int nb = cb ^ 1;
      if (ks < 3) {
#pragma unroll
        for (int m = 0; m < 4; ++m)
          af[nb][m] = load_fragA(hb + (m * 16) * 128 + (ks + 1) * 32, 128, lane);
#pragma unroll
        for (int i = 0; i < 2; ++i)
          bf_[nb][i] =
              load_fragB(wb + ((wave * 2 + i) * 16) * 128 + (ks + 1) * 32, 128, lane);
      }
#pragma unroll
      for (int i = 0; i < 2; ++i)
#pragma unroll
        for (int m = 0; m < 4; ++m)
          acc[m][i] = wmma_bf16(af[cb][m], bf_[cb][i], acc[m][i]);
    }
    __syncthreads();  // done reading `cur`; it may be re-staged next iter
  }

  // Spill accumulators: g_s[b][N] where N = gate*64 + col
  {
    const int nloc = lane & 15;
    const int mh   = (lane >> 4) * 8;
#pragma unroll
    for (int i = 0; i < 2; ++i) {
      const int nt = wave * 2 + i;
#pragma unroll
      for (int m = 0; m < 4; ++m)
#pragma unroll
        for (int j = 0; j < 8; ++j)
          g_s[(m * 16 + mh + j) * 256 + nt * 16 + nloc] = acc[m][i][j];
    }
  }
  __syncthreads();

  // Pointwise LSTM update for the block's 64 columns x 64 batch rows
  for (int e = tid; e < 64 * 64; e += 256) {
    const int cc = e & 63;
    const int b  = e >> 6;
    const float* pb = P_t + (size_t)b * G4;
    const float* gb = g_s + b * 256;
    float iv = sigmoidf_(gb[cc]       + pb[0 * HID + n0 + cc]);
    float fv = sigmoidf_(gb[64 + cc]  + pb[1 * HID + n0 + cc]);
    float gv = tanhf    (gb[128 + cc] + pb[2 * HID + n0 + cc]);
    float ov = sigmoidf_(gb[192 + cc] + pb[3 * HID + n0 + cc]);
    const size_t ci = (size_t)b * HID + n0 + cc;
    float cnew = fv * cst[ci] + iv * gv;
    cst[ci] = cnew;
    h_out[ci] = (bf16)(ov * tanhf(cnew));
  }
}

// ---------------------------------------------------------------------------
// Tail MLP: a = PReLU(additional @ fcc_W.T + fcc_b); out = [dh, a] @ fc_W.T + fc_b
// Grid: 64 (one block per batch row). Block: 256 threads.
// ---------------------------------------------------------------------------
__global__ void tail_kernel(const float* __restrict__ additional, // [64][128]
                            const float* __restrict__ fcc_W,      // [256][128]
                            const float* __restrict__ fcc_b,      // [256]
                            const float* __restrict__ fc_W,       // [1280]
                            const float* __restrict__ fc_b,       // [1]
                            const float* __restrict__ prelu_a,    // [1]
                            const bf16* __restrict__ dh,          // [64][1024]
                            float* __restrict__ out)              // [64]
{
  __shared__ float red[256];
  const int b   = blockIdx.x;
  const int tid = threadIdx.x;
  const float alpha = prelu_a[0];

  float aj = fcc_b[tid];
#pragma unroll 4
  for (int k = 0; k < 128; ++k)
    aj += additional[b * 128 + k] * fcc_W[tid * 128 + k];
  aj = (aj >= 0.0f) ? aj : alpha * aj;

  float partial = aj * fc_W[HID + tid];
#pragma unroll
  for (int j = tid; j < HID; j += 256)
    partial += (float)dh[(size_t)b * HID + j] * fc_W[j];

  red[tid] = partial;
  __syncthreads();
  for (int s = 128; s > 0; s >>= 1) {
    if (tid < s) red[tid] += red[tid + s];
    __syncthreads();
  }
  if (tid == 0) out[b] = red[0] + fc_b[0];
}

// ---------------------------------------------------------------------------
// Host driver
// ---------------------------------------------------------------------------
extern "C" void kernel_launch(void* const* d_in, const int* in_sizes, int n_in,
                              void* d_out, int out_size, void* d_ws, size_t ws_size,
                              hipStream_t stream) {
  const float* x        = (const float*)d_in[0];
  const float* addl     = (const float*)d_in[1];
  const float* enc_Wih  = (const float*)d_in[2];
  const float* enc_Whh  = (const float*)d_in[3];
  const float* enc_bih  = (const float*)d_in[4];
  const float* enc_bhh  = (const float*)d_in[5];
  const float* dec_Wih  = (const float*)d_in[6];
  const float* dec_Whh  = (const float*)d_in[7];
  const float* dec_bih  = (const float*)d_in[8];
  const float* dec_bhh  = (const float*)d_in[9];
  const float* fcc_W    = (const float*)d_in[10];
  const float* fcc_b    = (const float*)d_in[11];
  const float* fc_W     = (const float*)d_in[12];
  const float* fc_b     = (const float*)d_in[13];
  const float* prelu_a  = (const float*)d_in[14];

  // Workspace layout
  char* ws = (char*)d_ws;
  size_t off = 0;
  float* P = (float*)(ws + off);        off += (size_t)SEQT * BATCH * G4 * 4;  // 268 MB
  bf16* xb    = (bf16*)(ws + off);      off += (size_t)BATCH * SEQT * INP * 2; // 8 MB
  bf16* wih_e = (bf16*)(ws + off);      off += (size_t)G4 * INP * 2;           // 2 MB
  bf16* whh_e = (bf16*)(ws + off);      off += (size_t)G4 * HID * 2;           // 8 MB
  bf16* wih_d = (bf16*)(ws + off);      off += (size_t)G4 * INP * 2;
  bf16* whh_d = (bf16*)(ws + off);      off += (size_t)G4 * HID * 2;
  bf16* h_a   = (bf16*)(ws + off);      off += (size_t)BATCH * HID * 2;
  bf16* h_b   = (bf16*)(ws + off);      off += (size_t)BATCH * HID * 2;
  float* cbuf = (float*)(ws + off);     off += (size_t)BATCH * HID * 4;

  // --- Convert weights/activations to bf16 ---
  {
    int n;
    n = BATCH * SEQT * INP;
    cvt_f32_to_bf16<<<(n + 255) / 256, 256, 0, stream>>>(x, xb, n);
    n = G4 * INP;
    cvt_f32_to_bf16<<<(n + 255) / 256, 256, 0, stream>>>(enc_Wih, wih_e, n);
    cvt_f32_to_bf16<<<(n + 255) / 256, 256, 0, stream>>>(dec_Wih, wih_d, n);
    n = G4 * HID;
    cvt_f32_to_bf16<<<(n + 255) / 256, 256, 0, stream>>>(enc_Whh, whh_e, n);
    cvt_f32_to_bf16<<<(n + 255) / 256, 256, 0, stream>>>(dec_Whh, whh_d, n);
  }
  // --- Zero initial h and c ---
  {
    int n = BATCH * HID * 2 / 4;  // h_a as u32
    zero_u32<<<(n + 255) / 256, 256, 0, stream>>>((unsigned int*)h_a, n);
    n = BATCH * HID;              // cbuf as u32
    zero_u32<<<(n + 255) / 256, 256, 0, stream>>>((unsigned int*)cbuf, n);
  }

  const size_t smem_xproj = (size_t)BATCH * INP * 2 + (size_t)128 * INP * 2;  // 96 KB
  const size_t smem_step  = 2 * (HCHUNK_B + WCHUNK_B) + (size_t)BATCH * 256 * 4; // 224 KB
  dim3 gp(SEQT, G4 / 128);

  // --- Encoder ---
  xproj_kernel<<<gp, 256, smem_xproj, stream>>>(xb, wih_e, enc_bih, enc_bhh, P);
  for (int t = 0; t < SEQT; ++t) {
    const bf16* hi = (t & 1) ? h_b : h_a;
    bf16*       ho = (t & 1) ? h_a : h_b;
    lstm_step_kernel<<<HID / 64, 256, smem_step, stream>>>(
        hi, ho, cbuf, whh_e, P + (size_t)t * BATCH * G4);
  }
  // t=255 wrote into h_a -> encoder final h in h_a, c_T in cbuf.

  // --- Decoder (starts from encoder final state) ---
  xproj_kernel<<<gp, 256, smem_xproj, stream>>>(xb, wih_d, dec_bih, dec_bhh, P);
  for (int t = 0; t < SEQT; ++t) {
    const bf16* hi = (t & 1) ? h_b : h_a;
    bf16*       ho = (t & 1) ? h_a : h_b;
    lstm_step_kernel<<<HID / 64, 256, smem_step, stream>>>(
        hi, ho, cbuf, whh_d, P + (size_t)t * BATCH * G4);
  }
  // decoder final h in h_a.

  // --- Tail MLP ---
  tail_kernel<<<BATCH, 256, 0, stream>>>(addl, fcc_W, fcc_b, fc_W, fc_b, prelu_a,
                                         h_a, (float*)d_out);
}
